// LGMLoss_62852551410006
// MI455X (gfx1250) — compile-verified, matrix operand order checked by reading, not run on hardware
//
#include <hip/hip_runtime.h>

#define B_ROWS 4096
#define C_CLS  32000
#define D_DIM  128
#define ALPHA_HALF 0.05f   // ALPHA/2

typedef __attribute__((ext_vector_type(16))) __bf16 v16bf;
typedef __attribute__((ext_vector_type(8)))  float  v8f;
typedef __attribute__((ext_vector_type(4)))  unsigned int v4u;
typedef __attribute__((ext_vector_type(8)))  int v8i_;
typedef __attribute__((ext_vector_type(4)))  int v4i_;

struct Frag32 { uint4 a, b; };  // 32 bytes = one v16bf fragment

__device__ __forceinline__ unsigned short f2bf(float x) {
  unsigned u = __builtin_bit_cast(unsigned, x);
  u += 0x7FFFu + ((u >> 16) & 1u);          // round-to-nearest-even
  return (unsigned short)(u >> 16);
}
__device__ __forceinline__ float bf2f(unsigned short h) {
  unsigned u = ((unsigned)h) << 16;
  return __builtin_bit_cast(float, u);
}
// 16-bit A/B fragment K index for element e (0..15) in lane (ISA 7.12.2, 16-bit 16x32)
__device__ __forceinline__ int kmap(int e, int lane) {
  return (e & 7) + ((lane & 16) ? 8 : 0) + ((e & 8) ? 16 : 0);
}
__device__ __forceinline__ v16bf load_frag_lds(const unsigned short* p) {
  Frag32 f = *reinterpret_cast<const Frag32*>(p);
  return __builtin_bit_cast(v16bf, f);
}

// ---- 1-D TDM copy global -> LDS (data_size = 8B, single row of n8 elements).
// D# layout per CDNA5 ISA 8.3/8.4: group0 = {flags, lds_addr, global_addr, type},
// group1 = {data_size, tensor_dim0, tile_dim0, stride}; groups 2/3 zero (<=2D).
// This toolchain exposes the 6-arg builtin:
//   (uint32x4 g0, int32x8 g1, int32x4 g2, int32x4 g3, int32x8 g4, i32 cpol)
__device__ __forceinline__ void tdm_copy_1d(const void* gsrc, unsigned lds_off,
                                            unsigned nbytes) {
  unsigned long long ga = (unsigned long long)gsrc;
  unsigned n8 = nbytes >> 3;
  v4u g0;
  g0[0] = 1u;                                          // count=1, user mode
  g0[1] = lds_off;                                     // lds_addr (bytes)
  g0[2] = (unsigned)(ga & 0xFFFFFFFFu);                // global_addr[31:0]
  g0[3] = (unsigned)((ga >> 32) & 0x01FFFFFFu) | (2u << 30);  // addr[56:32] | type=2
  v8i_ g1;
  g1[0] = (int)(3u << 16);                 // workgroup_mask=0, data_size=3 (8B)
  g1[1] = (int)((n8 & 0xFFFFu) << 16);     // tensor_dim0[15:0] (bits 63:48)
  g1[2] = (int)((n8 >> 16) & 0xFFFFu);     // tensor_dim0[31:16]; tensor_dim1=0
  g1[3] = (int)((n8 & 0xFFFFu) << 16);     // tile_dim0 = n8 (bits 127:112)
  g1[4] = 0;                               // tile_dim1=0, tile_dim2=0 (unused)
  g1[5] = (int)n8;                         // tensor_dim0_stride[31:0]
  g1[6] = 0;
  g1[7] = 0;
  v4i_ z4 = {0, 0, 0, 0};
  v8i_ z8 = {0, 0, 0, 0, 0, 0, 0, 0};
  __builtin_amdgcn_tensor_load_to_lds(g0, g1, z4, z4, z8, 0);
}

// ---- prep A: A[b, 0:128] = feat^2, A[b,128:256] = feat, split to bf16 hi/lo,
// ---- stored pre-swizzled in WMMA A-fragment order: [(mtile*8+ks)*32+lane][e]
__global__ __launch_bounds__(256) void prep_A(const float* __restrict__ feat,
                                              unsigned short* __restrict__ Ahi,
                                              unsigned short* __restrict__ Alo) {
  int tid  = blockIdx.x * 256 + threadIdx.x;   // (mtile*8 + ks)*32 + lane
  int lane = tid & 31;
  int ks   = (tid >> 5) & 7;
  int m    = (tid >> 8) * 16 + (lane & 15);
  size_t base = (size_t)tid * 16;
  #pragma unroll
  for (int e = 0; e < 16; ++e) {
    int k = ks * 32 + kmap(e, lane);
    float v;
    if (k < D_DIM) { float f = feat[m * D_DIM + k]; v = f * f; }
    else           { v = feat[m * D_DIM + (k - D_DIM)]; }
    unsigned short h = f2bf(v);
    Ahi[base + e] = h;
    Alo[base + e] = f2bf(v - bf2f(h));
  }
}

// ---- prep W: W[c, 0:128] = exp(-lc), W[c,128:256] = -2*mu*exp(-lc), bf16 hi/lo,
// ---- stored pre-swizzled in WMMA B-fragment order
__global__ __launch_bounds__(256) void prep_W(const float* __restrict__ centers,
                                              const float* __restrict__ log_covs,
                                              unsigned short* __restrict__ Whi,
                                              unsigned short* __restrict__ Wlo) {
  int tid  = blockIdx.x * 256 + threadIdx.x;   // (ntile*8 + ks)*32 + lane
  int lane = tid & 31;
  int ks   = (tid >> 5) & 7;
  int n    = (tid >> 8) * 16 + (lane & 15);
  size_t base = (size_t)tid * 16;
  #pragma unroll
  for (int e = 0; e < 16; ++e) {
    int k = ks * 32 + kmap(e, lane);
    float v;
    if (k < D_DIM) {
      v = __expf(-log_covs[(size_t)n * D_DIM + k]);
    } else {
      int d = k - D_DIM;
      float inv = __expf(-log_covs[(size_t)n * D_DIM + d]);
      v = -2.0f * centers[(size_t)n * D_DIM + d] * inv;
    }
    unsigned short h = f2bf(v);
    Whi[base + e] = h;
    Wlo[base + e] = f2bf(v - bf2f(h));
  }
}

// ---- per-class constants: kconst[c] = sum mu^2*inv, slog[c] = sum log_covs
__global__ __launch_bounds__(128) void prep_const(const float* __restrict__ centers,
                                                  const float* __restrict__ log_covs,
                                                  float* __restrict__ kconst,
                                                  float* __restrict__ slog) {
  int c = blockIdx.x, t = threadIdx.x;
  float lc = log_covs[(size_t)c * D_DIM + t];
  float mu = centers[(size_t)c * D_DIM + t];
  __shared__ float s1[128], s2[128];
  s1[t] = mu * mu * __expf(-lc);
  s2[t] = lc;
  __syncthreads();
  for (int off = 64; off > 0; off >>= 1) {
    if (t < off) { s1[t] += s1[t + off]; s2[t] += s2[t + off]; }
    __syncthreads();
  }
  if (t == 0) { kconst[c] = s1[0]; slog[c] = s2[0]; }
}

// ---- split-bf16 compensated GEMM + fused pre-logit epilogue, TDM-staged.
// Block = 8 waves (2 along M x 4 along N). Wave tile = 32x32 (4 accs).
// Block tile = 64x128, entire K=256 band staged in LDS by the Tensor Data Mover:
//   Ahi/Alo bands 32KB each, Whi/Wlo bands 64KB each -> 192KB dynamic LDS.
// Grid = (32000/128, 4096/64) = (250, 64).
__global__ __launch_bounds__(256) void gemm_logits(
    const unsigned short* __restrict__ Ahi, const unsigned short* __restrict__ Alo,
    const unsigned short* __restrict__ Whi, const unsigned short* __restrict__ Wlo,
    const float* __restrict__ kconst, const float* __restrict__ slog,
    const int* __restrict__ label, float* __restrict__ out) {
  const size_t BC = (size_t)B_ROWS * C_CLS;
  extern __shared__ unsigned short lds[];
  // ushort offsets inside LDS
  const unsigned LDS_AHI = 0;       // 16384 ushorts (32KB)
  const unsigned LDS_ALO = 16384;   // 16384
  const unsigned LDS_WHI = 32768;   // 32768 ushorts (64KB)
  const unsigned LDS_WLO = 65536;   // 32768

  int lane = threadIdx.x & 31;
  int wave = threadIdx.x >> 5;
  int wm = wave & 1, wn = wave >> 1;
  int mt0 = blockIdx.y * 4 + wm * 2;   // global 16-row tiles
  int nt0 = blockIdx.x * 8 + wn * 2;   // global 16-col tiles

  // Wave 0 DMAs the block's K-bands into LDS via the Tensor Data Mover.
  if (wave == 0) {
    unsigned lds_base = (unsigned)(unsigned long long)(const void*)lds;
    size_t abase = (size_t)blockIdx.y * 16384;   // 4 mtiles * 8 ks * 32 * 16 elems
    size_t wbase = (size_t)blockIdx.x * 32768;   // 8 ntiles * 8 ks * 32 * 16 elems
    tdm_copy_1d(Ahi + abase, lds_base + 2u * LDS_AHI, 32768u);
    tdm_copy_1d(Alo + abase, lds_base + 2u * LDS_ALO, 32768u);
    tdm_copy_1d(Whi + wbase, lds_base + 2u * LDS_WHI, 65536u);
    tdm_copy_1d(Wlo + wbase, lds_base + 2u * LDS_WLO, 65536u);
    __builtin_amdgcn_s_wait_tensorcnt(0);
  }
  __syncthreads();

  v8f zero = {0.f, 0.f, 0.f, 0.f, 0.f, 0.f, 0.f, 0.f};
  v8f acc[2][2] = {{zero, zero}, {zero, zero}};

  for (int ks = 0; ks < 8; ++ks) {     // K = 256 in steps of 32
    v16bf aH[2], aL[2], bH[2], bL[2];
    #pragma unroll
    for (int i = 0; i < 2; ++i) {
      unsigned fb = ((unsigned)((wm * 2 + i) * 8 + ks) * 32 + lane) * 16;
      aH[i] = load_frag_lds(lds + LDS_AHI + fb);
      aL[i] = load_frag_lds(lds + LDS_ALO + fb);
    }
    #pragma unroll
    for (int j = 0; j < 2; ++j) {
      unsigned fb = ((unsigned)((wn * 2 + j) * 8 + ks) * 32 + lane) * 16;
      bH[j] = load_frag_lds(lds + LDS_WHI + fb);
      bL[j] = load_frag_lds(lds + LDS_WLO + fb);
    }
    #pragma unroll
    for (int i = 0; i < 2; ++i)
      #pragma unroll
      for (int j = 0; j < 2; ++j) {
        acc[i][j] = __builtin_amdgcn_wmma_f32_16x16x32_bf16(false, aH[i], false, bH[j], (short)0, acc[i][j], false, false);
        acc[i][j] = __builtin_amdgcn_wmma_f32_16x16x32_bf16(false, aH[i], false, bL[j], (short)0, acc[i][j], false, false);
        acc[i][j] = __builtin_amdgcn_wmma_f32_16x16x32_bf16(false, aL[i], false, bH[j], (short)0, acc[i][j], false, false);
      }
  }

  #pragma unroll
  for (int i = 0; i < 2; ++i) {
    int mbase = (mt0 + i) * 16 + ((lane >> 4) & 1) * 8;   // C/D layout: M = r + 8*(lane>=16)
    #pragma unroll
    for (int j = 0; j < 2; ++j) {
      int n = (nt0 + j) * 16 + (lane & 15);               // C/D layout: N = lane&15
      float kc = kconst[n], sl = slog[n];
      union { v8f v; float f[8]; } u; u.v = acc[i][j];
      #pragma unroll
      for (int r = 0; r < 8; ++r) {
        int m = mbase + r;
        float dist = u.f[r] + kc;
        float l = -0.5f * (sl + dist);
        size_t off = (size_t)m * C_CLS + n;
        out[off] = l;                                     // pre-logits
        float ml = (n == label[m]) ? (l - ALPHA_HALF * dist) : l;
        out[BC + off] = ml;                               // pre-margin-logits
      }
    }
  }
}

// ---- row-wise online log_softmax for both planes (deterministic fixed-order reduce)
__global__ __launch_bounds__(256) void softmax_rows(float* __restrict__ out) {
  const size_t BC = (size_t)B_ROWS * C_CLS;
  int row = blockIdx.x, t = threadIdx.x;
  float* L  = out + (size_t)row * C_CLS;
  float* Mg = out + BC + (size_t)row * C_CLS;
  float m1 = -3.4e38f, s1 = 0.f, m2 = -3.4e38f, s2 = 0.f;
  for (int c = t; c < C_CLS; c += 256) {
    float x = L[c];
    if (x > m1) { s1 = s1 * __expf(m1 - x) + 1.f; m1 = x; } else { s1 += __expf(x - m1); }
    float y = Mg[c];
    if (y > m2) { s2 = s2 * __expf(m2 - y) + 1.f; m2 = y; } else { s2 += __expf(y - m2); }
  }
  __shared__ float sm1[256], ss1[256], sm2[256], ss2[256];
  sm1[t] = m1; ss1[t] = s1; sm2[t] = m2; ss2[t] = s2;
  __syncthreads();
  for (int off = 128; off > 0; off >>= 1) {
    if (t < off) {
      float ma = sm1[t], mb = sm1[t + off], M = fmaxf(ma, mb);
      ss1[t] = ss1[t] * __expf(ma - M) + ss1[t + off] * __expf(mb - M);
      sm1[t] = M;
      ma = sm2[t]; mb = sm2[t + off]; M = fmaxf(ma, mb);
      ss2[t] = ss2[t] * __expf(ma - M) + ss2[t + off] * __expf(mb - M);
      sm2[t] = M;
    }
    __syncthreads();
  }
  __shared__ float lse1, lse2;
  if (t == 0) { lse1 = sm1[0] + __logf(ss1[0]); lse2 = sm2[0] + __logf(ss2[0]); }
  __syncthreads();
  float a = lse1, b = lse2;
  for (int c = t; c < C_CLS; c += 256) { L[c] -= a; Mg[c] -= b; }
}

// ---- per-row scalar partials: ||feat - mu_label||^2 and slog[label]
__global__ __launch_bounds__(128) void row_stats(const float* __restrict__ feat,
                                                 const int* __restrict__ label,
                                                 const float* __restrict__ centers,
                                                 const float* __restrict__ slog,
                                                 float* __restrict__ rowpart,
                                                 float* __restrict__ sloglab) {
  int row = blockIdx.x, t = threadIdx.x;
  int lab = label[row];
  float d = feat[(size_t)row * D_DIM + t] - centers[(size_t)lab * D_DIM + t];
  __shared__ float s[128];
  s[t] = d * d;
  __syncthreads();
  for (int off = 64; off > 0; off >>= 1) {
    if (t < off) s[t] += s[t + off];
    __syncthreads();
  }
  if (t == 0) { rowpart[row] = s[0]; sloglab[row] = slog[lab]; }
}

__global__ __launch_bounds__(256) void finalize(const float* __restrict__ rowpart,
                                                const float* __restrict__ sloglab,
                                                float* __restrict__ out) {
  const size_t BC2 = 2ull * B_ROWS * C_CLS;
  int t = threadIdx.x;
  float a = 0.f, b = 0.f;
  for (int i = t; i < B_ROWS; i += 256) { a += rowpart[i]; b += sloglab[i]; }
  __shared__ float sa[256], sb[256];
  sa[t] = a; sb[t] = b;
  __syncthreads();
  for (int off = 128; off > 0; off >>= 1) {
    if (t < off) { sa[t] += sa[t + off]; sb[t] += sb[t + off]; }
    __syncthreads();
  }
  if (t == 0) {
    float cdist = 0.5f * sa[0];
    float reg = 0.5f * (sb[0] + 1e-8f);
    float likelihood = (cdist - reg) / (float)B_ROWS;
    out[BC2 + 0] = likelihood;
    out[BC2 + 1] = cdist;
    out[BC2 + 2] = reg;
  }
}

extern "C" void kernel_launch(void* const* d_in, const int* in_sizes, int n_in,
                              void* d_out, int out_size, void* d_ws, size_t ws_size,
                              hipStream_t stream) {
  const float* feat     = (const float*)d_in[0];
  const int*   label    = (const int*)d_in[1];
  const float* centers  = (const float*)d_in[2];
  const float* log_covs = (const float*)d_in[3];
  float* out = (float*)d_out;
  char* ws = (char*)d_ws;

  // workspace layout (bytes), ~37.25 MB total
  unsigned short* Ahi = (unsigned short*)(ws + 0);          // 4096*256*2   = 2,097,152
  unsigned short* Alo = (unsigned short*)(ws + 2097152);    // 2,097,152
  unsigned short* Whi = (unsigned short*)(ws + 4194304);    // 32000*256*2  = 16,384,000
  unsigned short* Wlo = (unsigned short*)(ws + 20578304);   // 16,384,000
  float* kconst  = (float*)(ws + 36962304);                 // 128,000
  float* slog    = (float*)(ws + 37090304);                 // 128,000
  float* rowpart = (float*)(ws + 37218304);                 // 16,384
  float* sloglab = (float*)(ws + 37234688);                 // 16,384

  prep_const<<<C_CLS, 128, 0, stream>>>(centers, log_covs, kconst, slog);
  prep_A<<<256, 256, 0, stream>>>(feat, Ahi, Alo);          // 256*8*32 threads
  prep_W<<<2000, 256, 0, stream>>>(centers, log_covs, Whi, Wlo);
  gemm_logits<<<dim3(250, 64), 256, 196608, stream>>>(Ahi, Alo, Whi, Wlo,
                                                      kconst, slog, label, out);
  softmax_rows<<<B_ROWS, 256, 0, stream>>>(out);
  row_stats<<<B_ROWS, 128, 0, stream>>>(feat, label, centers, slog, rowpart, sloglab);
  finalize<<<1, 256, 0, stream>>>(rowpart, sloglab, out);
}
